// LMModel_16252156248502
// MI455X (gfx1250) — compile-verified, matrix operand order checked by reading, not run on hardware
//
#include <hip/hip_runtime.h>
#include <hip/hip_bf16.h>

// ---------------- model dims ----------------
#define T_    128
#define B_    8
#define NHID_ 512
#define H4_   2048
#define DH_   256
#define NVOC_ 32000
#define NIN_  512

// ---------------- vector types ----------------
typedef __attribute__((ext_vector_type(16))) __bf16 v16bf;
typedef __attribute__((ext_vector_type(8)))  __bf16 v8bf_t;
typedef __attribute__((ext_vector_type(8)))  float  v8f;
typedef __attribute__((ext_vector_type(4)))  unsigned int u32x4;
typedef __attribute__((ext_vector_type(8)))  int    i32x8;
typedef __attribute__((ext_vector_type(4)))  int    i32x4;

__device__ __forceinline__ v8f wmma_bf16(v16bf a, v16bf b, v8f c) {
  return __builtin_amdgcn_wmma_f32_16x16x32_bf16(
      false, a, false, b, (short)0, c, false, false);
}

// Load one 16x32 bf16 fragment (row-major source, row stride = ld elements).
// ISA 16-bit A layout: lanes 0-15 hold row M=lane, K = {0..7} U {16..23};
// lanes 16-31 hold row M=lane-16, K = {8..15} U {24..31}.
__device__ __forceinline__ v16bf load_frag(const __bf16* base, int ld) {
  int lane = threadIdx.x & 31;
  int row  = lane & 15;
  int kb   = (lane & 16) ? 8 : 0;
  const __bf16* p = base + (long)row * ld + kb;
  v8bf_t lo = *(const v8bf_t*)(p);
  v8bf_t hi = *(const v8bf_t*)(p + 16);
  return __builtin_shufflevector(lo, hi, 0,1,2,3,4,5,6,7,8,9,10,11,12,13,14,15);
}

__device__ __forceinline__ float sigmoidf_(float x) {
  return 1.f / (1.f + __expf(-x));
}

// ---------------- generic bf16 GEMM: C[M,N] = A[M,K] * W[N,K]^T + b1 + b2 ----
// block = 256 threads (8 waves, 2x4), block tile 32(M) x 256(N), wave tile 16x64
// K-loop unrolled x2 with ping-pong fragment sets: set1 loads are issued before
// set0's WMMAs (and vice versa) so loads overlap matrix ops without any
// register-rotation moves. Requires K % 64 == 0 (K is 256 or 512 here).
__global__ void gemm_bf16(const __bf16* __restrict__ A, int lda,
                          const __bf16* __restrict__ W, int ldw,
                          float* __restrict__ C, int ldc,
                          const float* __restrict__ bias1,
                          const float* __restrict__ bias2,
                          int K) {
  int n0   = blockIdx.x * 256;
  int m0   = blockIdx.y * 32;
  int wave = threadIdx.x >> 5;
  int lane = threadIdx.x & 31;
  int wm   = (wave >> 2) * 16;
  int wn   = (wave & 3) * 64;

  const __bf16* Arow  = A + (long)(m0 + wm) * lda;
  const __bf16* Wbase = W + (long)(n0 + wn) * ldw;

  v8f acc[4] = {};
  v16bf a0 = load_frag(Arow, lda);
  v16bf b0[4];
#pragma unroll
  for (int j = 0; j < 4; ++j) b0[j] = load_frag(Wbase + (long)j * 16 * ldw, ldw);

  for (int k = 0; k < K; k += 64) {
    // stage 1: issue loads for k+32 (always in-range: K % 64 == 0), compute k
    v16bf a1 = load_frag(Arow + k + 32, lda);
    v16bf b1[4];
#pragma unroll
    for (int j = 0; j < 4; ++j) b1[j] = load_frag(Wbase + (long)j * 16 * ldw + k + 32, ldw);
#pragma unroll
    for (int j = 0; j < 4; ++j) acc[j] = wmma_bf16(a0, b0[j], acc[j]);

    // stage 2: issue loads for k+64 (if any), compute k+32
    if (k + 64 < K) {
      __builtin_prefetch(Arow + k + 96, 0, 0);
      a0 = load_frag(Arow + k + 64, lda);
#pragma unroll
      for (int j = 0; j < 4; ++j) b0[j] = load_frag(Wbase + (long)j * 16 * ldw + k + 64, ldw);
    }
#pragma unroll
    for (int j = 0; j < 4; ++j) acc[j] = wmma_bf16(a1, b1[j], acc[j]);
  }

  int nin = lane & 15;
  int mb  = (lane < 16) ? 0 : 8;
#pragma unroll
  for (int j = 0; j < 4; ++j) {
    int n = n0 + wn + j * 16 + nin;
    float bv = (bias1 ? bias1[n] : 0.f) + (bias2 ? bias2[n] : 0.f);
#pragma unroll
    for (int r = 0; r < 8; ++r) {
      int m = m0 + wm + mb + r;
      C[(long)m * ldc + n] = acc[j][r] + bv;
    }
  }
}

// ---------------- decoder GEMM with TDM staging of W into LDS ---------------
// C[M,N] = A[M,512] * W[N,512]^T + bias, block tile 32x256.
// Wave 0 drives the Tensor Data Mover: per 32-wide k-chunk it DMAs a 256x32
// bf16 tile of W into a double-buffered LDS tile (tensor_load_to_lds,
// s_wait_tensorcnt), overlapped with WMMA on the other buffer.
// Output written with nontemporal stores (131MB stream, don't pollute L2).
__global__ void gemm_dec_tdm(const __bf16* __restrict__ A, int lda,
                             const __bf16* __restrict__ W,
                             const float* __restrict__ bias,
                             float* __restrict__ C, int ldc, int K) {
  __shared__ __align__(128) __bf16 wtile[2][256 * 32];
  int n0   = blockIdx.x * 256;
  int m0   = blockIdx.y * 32;
  int wave = threadIdx.x >> 5;
  int lane = threadIdx.x & 31;
  int wm   = (wave >> 2) * 16;
  int wn   = (wave & 3) * 64;
  const int KC = K / 32;

  // constant part of the tensor descriptor (D# group 1):
  //   data_size=2B, tensor_dim0=512, tensor_dim1=NVOC, tile=32x256, stride0=512
  const i32x8 g1 = { (int)(1u << 16),              // data_size = 1 (2 bytes)
                     (int)(512u << 16),            // tensor_dim0[15:0] @ bits63:48
                     (int)((unsigned)NVOC_ << 16), // tensor_dim1[15:0] @ bits95:80
                     (int)(32u << 16),             // tile_dim0 @ bits127:112
                     256,                          // tile_dim1 @ bits143:128
                     512,                          // tensor_dim0_stride[31:0]
                     0, 0 };
  const i32x4 gz4 = { 0, 0, 0, 0 };
  const i32x8 gz8 = { 0, 0, 0, 0, 0, 0, 0, 0 };

  auto issue_tdm = [&](int kc, int buf) {
    unsigned long long ga = (unsigned long long)(size_t)W +
                            ((unsigned long long)n0 * (unsigned)K + (unsigned)(kc * 32)) * 2ull;
    unsigned int lds_base = (unsigned int)(size_t)(&wtile[buf][0]);
    u32x4 g0 = { 1u,                              // count=1, is_restore=0
                 lds_base,                        // lds_addr @ bits63:32
                 (unsigned int)(ga & 0xffffffffu),
                 (unsigned int)(ga >> 32) | 0x80000000u };  // type=2 @ bits127:126
    __builtin_amdgcn_tensor_load_to_lds(g0, g1, gz4, gz4, gz8, 0);
  };

  if (wave == 0) {
    issue_tdm(0, 0);
    __builtin_amdgcn_s_wait_tensorcnt(0);
  }
  __syncthreads();

  v8f acc[4] = {};
  for (int kc = 0; kc < KC; ++kc) {
    int buf = kc & 1;
    if (wave == 0 && kc + 1 < KC) issue_tdm(kc + 1, buf ^ 1);   // overlap DMA
    v16bf a = load_frag(A + (long)(m0 + wm) * lda + kc * 32, lda);
#pragma unroll
    for (int j = 0; j < 4; ++j) {
      v16bf b = load_frag(&wtile[buf][(wn + j * 16) * 32], 32);
      acc[j] = wmma_bf16(a, b, acc[j]);
    }
    if (wave == 0 && kc + 1 < KC) __builtin_amdgcn_s_wait_tensorcnt(0);
    __syncthreads();
  }

  int nin = lane & 15;
  int mb  = (lane < 16) ? 0 : 8;
#pragma unroll
  for (int j = 0; j < 4; ++j) {
    int n = n0 + wn + j * 16 + nin;
    float bv = bias ? bias[n] : 0.f;
#pragma unroll
    for (int r = 0; r < 8; ++r) {
      int m = m0 + wm + mb + r;
      __builtin_nontemporal_store(acc[j][r] + bv, &C[(long)m * ldc + n]);
    }
  }
}

// ---------------- elementwise helpers ----------------
__global__ void f2bf_kernel(const float* __restrict__ src, __bf16* __restrict__ dst, long n) {
  long i = (long)blockIdx.x * blockDim.x + threadIdx.x;
  long s = (long)gridDim.x * blockDim.x;
  for (; i < n; i += s) dst[i] = (__bf16)src[i];
}

__global__ void embed_kernel(const int* __restrict__ tokens, const float* __restrict__ emb,
                             float* __restrict__ x, __bf16* __restrict__ x_bf) {
  long idx = (long)blockIdx.x * blockDim.x + threadIdx.x;   // T*B*NIN
  int row = (int)(idx >> 9);
  int col = (int)(idx & 511);
  int tok = tokens[row];
  float v = emb[(long)tok * NIN_ + col];
  x[idx] = v;
  x_bf[idx] = (__bf16)v;
}

// ---------------- LSTM layer scan (one workgroup, 8 waves) ----------------
__global__ void lstm_scan(const float* __restrict__ xW, const __bf16* __restrict__ Whh,
                          float* __restrict__ hs, __bf16* __restrict__ hs_bf) {
  __shared__ __bf16 h_bf[16][NHID_];   // rows 8..15 stay zero (M padding)
  __shared__ float  c_st[8][NHID_];
  __shared__ float  g_st[8][H4_];
  int tid = threadIdx.x;
  for (int i = tid; i < 16 * NHID_; i += 256) (&h_bf[0][0])[i] = (__bf16)0.f;
  for (int i = tid; i < 8 * NHID_; i += 256) (&c_st[0][0])[i] = 0.f;
  __syncthreads();
  int wave = tid >> 5, lane = tid & 31;

  for (int t = 0; t < T_; ++t) {
    for (int nt = 0; nt < 16; ++nt) {
      int n0 = wave * 256 + nt * 16;
      v8f acc = {};
      for (int k = 0; k < NHID_; k += 32) {
        v16bf a = load_frag(&h_bf[0][0] + k, NHID_);
        v16bf b = load_frag(Whh + (long)n0 * NHID_ + k, NHID_);
        acc = wmma_bf16(a, b, acc);
      }
      if (lane < 16) {
        int n = n0 + (lane & 15);
#pragma unroll
        for (int r = 0; r < 8; ++r) g_st[r][n] = acc[r];
      }
    }
    __syncthreads();
    for (int i = tid; i < 8 * NHID_; i += 256) {
      int b = i >> 9, n = i & 511;
      const float* xw = xW + (long)(t * B_ + b) * H4_;
      float gi = g_st[b][n]              + xw[n];
      float gf = g_st[b][NHID_ + n]      + xw[NHID_ + n];
      float gg = g_st[b][2 * NHID_ + n]  + xw[2 * NHID_ + n];
      float go = g_st[b][3 * NHID_ + n]  + xw[3 * NHID_ + n];
      float cn = sigmoidf_(gf) * c_st[b][n] + sigmoidf_(gi) * tanhf(gg);
      float hn = sigmoidf_(go) * tanhf(cn);
      c_st[b][n] = cn;
      h_bf[b][n] = (__bf16)hn;
      long o = (long)(t * B_ + b) * NHID_ + n;
      hs[o] = hn;
      hs_bf[o] = (__bf16)hn;
    }
    __syncthreads();
  }
}

// ---------------- forward RNN scan (one workgroup) ----------------
__global__ void rnn_fwd_scan(const float* __restrict__ xW, const __bf16* __restrict__ Whh,
                             float* __restrict__ hf, __bf16* __restrict__ hf_bf) {
  __shared__ __bf16 h_bf[16][DH_];
  __shared__ float  g_st[8][DH_];
  int tid = threadIdx.x;
  for (int i = tid; i < 16 * DH_; i += 256) (&h_bf[0][0])[i] = (__bf16)0.f;
  __syncthreads();
  int wave = tid >> 5, lane = tid & 31;
  for (int t = 0; t < T_; ++t) {
    for (int nt = 0; nt < 2; ++nt) {
      int n0 = wave * 32 + nt * 16;
      v8f acc = {};
      for (int k = 0; k < DH_; k += 32) {
        v16bf a = load_frag(&h_bf[0][0] + k, DH_);
        v16bf b = load_frag(Whh + (long)n0 * DH_ + k, DH_);
        acc = wmma_bf16(a, b, acc);
      }
      if (lane < 16) {
        int n = n0 + (lane & 15);
#pragma unroll
        for (int r = 0; r < 8; ++r) g_st[r][n] = acc[r];
      }
    }
    __syncthreads();
    for (int i = tid; i < 8 * DH_; i += 256) {
      int b = i >> 8, n = i & 255;
      float hv = tanhf(g_st[b][n] + xW[(long)(t * B_ + b) * DH_ + n]);
      h_bf[b][n] = (__bf16)hv;
      long o = (long)(t * B_ + b) * DH_ + n;
      hf[o] = hv;
      hf_bf[o] = (__bf16)hv;
    }
    __syncthreads();
  }
}

// ---------------- backward per-prefix RNN: 128 independent scans ----------
__global__ void rnn_bwd_scan(const float* __restrict__ xWb, const __bf16* __restrict__ Whh,
                             __bf16* __restrict__ h_bwd_bf) {
  int i = blockIdx.x;
  __shared__ __bf16 h_bf[16][DH_];
  __shared__ float  g_st[8][DH_];
  int tid = threadIdx.x;
  for (int k = tid; k < 16 * DH_; k += 256) (&h_bf[0][0])[k] = (__bf16)0.f;
  __syncthreads();
  int wave = tid >> 5, lane = tid & 31;
  for (int t = i; t >= 0; --t) {
    for (int nt = 0; nt < 2; ++nt) {
      int n0 = wave * 32 + nt * 16;
      v8f acc = {};
      for (int k = 0; k < DH_; k += 32) {
        v16bf a = load_frag(&h_bf[0][0] + k, DH_);
        v16bf b = load_frag(Whh + (long)n0 * DH_ + k, DH_);
        acc = wmma_bf16(a, b, acc);
      }
      if (lane < 16) {
        int n = n0 + (lane & 15);
#pragma unroll
        for (int r = 0; r < 8; ++r) g_st[r][n] = acc[r];
      }
    }
    __syncthreads();
    for (int k = tid; k < 8 * DH_; k += 256) {
      int b = k >> 8, n = k & 255;
      float hv = tanhf(g_st[b][n] + xWb[(long)(t * B_ + b) * DH_ + n]);
      h_bf[b][n] = (__bf16)hv;
      h_bwd_bf[((long)(i * T_ + t) * B_ + b) * DH_ + n] = (__bf16)hv;
    }
    __syncthreads();
  }
}

// ---------------- fused attention scores (lower triangle only) --------------
__global__ void att_e_kernel(const __bf16* __restrict__ h_bwd_bf,
                             const __bf16* __restrict__ attW1_bf,   // (512,1024)
                             const float* __restrict__ s1, const float* __restrict__ f1,
                             const float* __restrict__ att_b1, const float* __restrict__ att_W2,
                             const float* __restrict__ att_b2,
                             float* __restrict__ e) {
  int i = blockIdx.x;
  int tbase = blockIdx.y * 16;
  if (tbase > i) return;
  int wave = threadIdx.x >> 5, lane = threadIdx.x & 31;
  int trow = tbase + wave * 2;
  const __bf16* Arows = h_bwd_bf + (long)(i * T_ + trow) * B_ * DH_;  // 16 rows, stride DH_
  int tloc = (lane < 16) ? 0 : 1;
  int t = trow + tloc;

  float accsum[8] = {};
  for (int ntile = 0; ntile < NIN_ / 16; ++ntile) {
    int n0 = ntile * 16;
    v8f c = {};
    for (int k = 0; k < DH_; k += 32) {
      v16bf a = load_frag(Arows + k, DH_);
      v16bf b = load_frag(attW1_bf + (long)n0 * (2 * NIN_) + (NHID_ + DH_) + k, 2 * NIN_);
      c = wmma_bf16(a, b, c);
    }
    int n = n0 + (lane & 15);
    float b1v = att_b1[n];
    float w2n = att_W2[n];
#pragma unroll
    for (int r = 0; r < 8; ++r) {
      int b = r;   // row m = tloc*8 + r  ->  b = r
      float pre = tanhf(c[r] + s1[(long)(i * B_ + b) * NIN_ + n]
                             + f1[(long)(t * B_ + b) * NIN_ + n] + b1v);
      accsum[r] += w2n * pre;
    }
  }
#pragma unroll
  for (int r = 0; r < 8; ++r) {
    float v = accsum[r];
    for (int off = 8; off >= 1; off >>= 1) v += __shfl_xor(v, off, 16);
    if ((lane & 15) == 0 && t <= i)
      e[(long)(i * T_ + t) * B_ + r] = tanhf(v + att_b2[0]);
  }
}

// ---------------- masked softmax over t<=i + context + residual -------------
__global__ void softmax_ctx(const float* __restrict__ e, const float* __restrict__ h_fwd,
                            const __bf16* __restrict__ h_bwd_bf, const float* __restrict__ outh,
                            __bf16* __restrict__ res_bf) {
  int i = blockIdx.x;
  __shared__ float a_sh[T_];
  __shared__ float red[256];
  int tid = threadIdx.x;
  int n = i + 1;  // valid t count
  for (int b = 0; b < B_; ++b) {
    float m = -1e30f;
    for (int t = tid; t < n; t += 256) m = fmaxf(m, e[(long)(i * T_ + t) * B_ + b]);
    red[tid] = m; __syncthreads();
    for (int s = 128; s >= 1; s >>= 1) {
      if (tid < s) red[tid] = fmaxf(red[tid], red[tid + s]);
      __syncthreads();
    }
    m = red[0]; __syncthreads();
    float sm = 0.f;
    for (int t = tid; t < n; t += 256) {
      float v = __expf(e[(long)(i * T_ + t) * B_ + b] - m);
      a_sh[t] = v; sm += v;
    }
    red[tid] = sm; __syncthreads();
    for (int s = 128; s >= 1; s >>= 1) {
      if (tid < s) red[tid] += red[tid + s];
      __syncthreads();
    }
    float inv = 1.f / red[0]; __syncthreads();

    int f = tid;                      // 256 threads == DH_ columns
    float cf = 0.f, cb = 0.f;
    for (int t = 0; t < n; ++t) {
      float at = a_sh[t] * inv;
      cf += at * h_fwd[(long)(t * B_ + b) * DH_ + f];
      cb += at * (float)h_bwd_bf[((long)(i * T_ + t) * B_ + b) * DH_ + f];
    }
    long row = (long)(i * B_ + b) * NHID_;
    res_bf[row + f]       = (__bf16)(cf + outh[row + f]);
    res_bf[row + DH_ + f] = (__bf16)(cb + outh[row + DH_ + f]);
    __syncthreads();
  }
}

// ---------------- host side ----------------
extern "C" void kernel_launch(void* const* d_in, const int* in_sizes, int n_in,
                              void* d_out, int out_size, void* d_ws, size_t ws_size,
                              hipStream_t stream) {
  (void)in_sizes; (void)n_in; (void)out_size; (void)ws_size;
  const int*   tokens    = (const int*)  d_in[0];
  const float* emb       = (const float*)d_in[1];
  const float* lstm_Wih  = (const float*)d_in[2];
  const float* lstm_Whh  = (const float*)d_in[3];
  const float* lstm_bih  = (const float*)d_in[4];
  const float* lstm_bhh  = (const float*)d_in[5];
  const float* rnn_Wih_f = (const float*)d_in[6];
  const float* rnn_Whh_f = (const float*)d_in[7];
  const float* rnn_b_f   = (const float*)d_in[8];
  const float* rnn_Wih_b = (const float*)d_in[9];
  const float* rnn_Whh_b = (const float*)d_in[10];
  const float* rnn_b_b   = (const float*)d_in[11];
  const float* att_W1    = (const float*)d_in[12];
  const float* att_b1    = (const float*)d_in[13];
  const float* att_W2    = (const float*)d_in[14];
  const float* att_b2    = (const float*)d_in[15];
  const float* dec_W     = (const float*)d_in[16];
  const float* dec_b     = (const float*)d_in[17];
  float* out = (float*)d_out;

  char* ws = (char*)d_ws;
  size_t off = 0;
  auto alloc = [&](size_t bytes) -> void* {
    off = (off + 255) & ~(size_t)255;
    void* p = ws + off;
    off += bytes;
    return p;
  };
  const long TB = (long)T_ * B_;  // 1024
  float*  x_f32    = (float*) alloc(TB * NIN_ * 4);
  __bf16* x_bf     = (__bf16*)alloc(TB * NIN_ * 2);
  float*  xW       = (float*) alloc(TB * H4_ * 4);
  float*  hs       = (float*) alloc(TB * NHID_ * 4);
  __bf16* hs_bf    = (__bf16*)alloc(TB * NHID_ * 2);
  float*  xWf      = (float*) alloc(TB * DH_ * 4);
  float*  xWb      = (float*) alloc(TB * DH_ * 4);
  float*  hfwd     = (float*) alloc(TB * DH_ * 4);
  __bf16* hfwd_bf  = (__bf16*)alloc(TB * DH_ * 2);
  __bf16* hbwd_bf  = (__bf16*)alloc((long)T_ * T_ * B_ * DH_ * 2);
  float*  s1       = (float*) alloc(TB * NIN_ * 4);
  float*  f1       = (float*) alloc(TB * NIN_ * 4);
  float*  e_buf    = (float*) alloc((long)T_ * T_ * B_ * 4);
  __bf16* res_bf   = (__bf16*)alloc(TB * NHID_ * 2);
  __bf16* Wih_bf   = (__bf16*)alloc(2L * H4_ * NHID_ * 2);
  __bf16* Whh_bf   = (__bf16*)alloc(2L * H4_ * NHID_ * 2);
  __bf16* Wihf_bf  = (__bf16*)alloc((long)DH_ * NIN_ * 2);
  __bf16* Whhf_bf  = (__bf16*)alloc((long)DH_ * DH_ * 2);
  __bf16* Wihb_bf  = (__bf16*)alloc((long)DH_ * NIN_ * 2);
  __bf16* Whhb_bf  = (__bf16*)alloc((long)DH_ * DH_ * 2);
  __bf16* attW1_bf = (__bf16*)alloc((long)NIN_ * 2 * NIN_ * 2);
  __bf16* decW_bf  = (__bf16*)alloc((long)NVOC_ * NHID_ * 2);

  dim3 blk(256);
  auto cvt = [&](const float* s, __bf16* d, long nelem) {
    int g = (int)((nelem + 255) / 256); if (g > 4096) g = 4096;
    f2bf_kernel<<<g, blk, 0, stream>>>(s, d, nelem);
  };
  cvt(lstm_Wih,  Wih_bf,   2L * H4_ * NHID_);
  cvt(lstm_Whh,  Whh_bf,   2L * H4_ * NHID_);
  cvt(rnn_Wih_f, Wihf_bf,  (long)DH_ * NIN_);
  cvt(rnn_Whh_f, Whhf_bf,  (long)DH_ * DH_);
  cvt(rnn_Wih_b, Wihb_bf,  (long)DH_ * NIN_);
  cvt(rnn_Whh_b, Whhb_bf,  (long)DH_ * DH_);
  cvt(att_W1,    attW1_bf, (long)NIN_ * 2 * NIN_);
  cvt(dec_W,     decW_bf,  (long)NVOC_ * NHID_);

  embed_kernel<<<(int)(TB * NIN_ / 256), blk, 0, stream>>>(tokens, emb, x_f32, x_bf);

  auto gemm = [&](const __bf16* A, int lda, const __bf16* W, int ldw,
                  float* C, int ldc, const float* b1, const float* b2,
                  int M, int N, int K) {
    dim3 g(N / 256, M / 32);
    gemm_bf16<<<g, blk, 0, stream>>>(A, lda, W, ldw, C, ldc, b1, b2, K);
  };

  // LSTM layers
  const __bf16* lstm_in = x_bf;
  for (int l = 0; l < 2; ++l) {
    gemm(lstm_in, NHID_, Wih_bf + (long)l * H4_ * NHID_, NHID_,
         xW, H4_, lstm_bih + l * H4_, lstm_bhh + l * H4_, (int)TB, H4_, NHID_);
    lstm_scan<<<1, blk, 0, stream>>>(xW, Whh_bf + (long)l * H4_ * NHID_, hs, hs_bf);
    lstm_in = hs_bf;
  }

  // RNN input projections
  gemm(x_bf, NIN_, Wihf_bf, NIN_, xWf, DH_, rnn_b_f, nullptr, (int)TB, DH_, NIN_);
  gemm(x_bf, NIN_, Wihb_bf, NIN_, xWb, DH_, rnn_b_b, nullptr, (int)TB, DH_, NIN_);

  // scans
  rnn_fwd_scan<<<1, blk, 0, stream>>>(xWf, Whhf_bf, hfwd, hfwd_bf);
  rnn_bwd_scan<<<T_, blk, 0, stream>>>(xWb, Whhb_bf, hbwd_bf);

  // attention precomputations: s1 = out@W1s^T, f1 = h_fwd@W1hf^T
  gemm(hs_bf, NHID_, attW1_bf, 2 * NIN_, s1, NIN_, nullptr, nullptr, (int)TB, NIN_, NHID_);
  gemm(hfwd_bf, DH_, attW1_bf + NHID_, 2 * NIN_, f1, NIN_, nullptr, nullptr, (int)TB, NIN_, DH_);

  // fused scores (lower triangle only)
  att_e_kernel<<<dim3(T_, T_ / 16), blk, 0, stream>>>(
      hbwd_bf, attW1_bf, s1, f1, att_b1, att_W2, att_b2, e_buf);

  // softmax + context + residual -> res_bf
  softmax_ctx<<<T_, blk, 0, stream>>>(e_buf, hfwd, hbwd_bf, hs, res_bf);

  // decoder: out = res @ dec_W^T + dec_b  (1024 x 32000), TDM-staged W
  {
    dim3 g(NVOC_ / 256, (int)TB / 32);
    gemm_dec_tdm<<<g, blk, 0, stream>>>(res_bf, NHID_, decW_bf, dec_b, out, NVOC_, NHID_);
  }
}
